// GCNmodel_9311489098373
// MI455X (gfx1250) — matrix operand order repaired and checked by reading, not compiled, not run on hardware
//
#include <hip/hip_runtime.h>
#include <math.h>

// ---------------- model constants ----------------
#define N_NODES 100000
#define N_EDGES 400000
#define N_GRAPHS 500
#define NPG 200          // nodes per graph
#define HD 128           // hidden
#define NHEADS 8
#define DH 16            // head dim
#define KSEEDS 75

// ---------------- WMMA tile config ----------------
#define WPB 4            // waves per block (wave32)
#define BM (16 * WPB)    // 64 rows per block
#define BK 32            // K per WMMA step (bf16 16x16x32)
#define BN 128           // output columns (always 128 here)
#define BKP 40           // padded LDS row in bf16 (80 B: 16B-aligned chunks, bank-conflict-free)

typedef __attribute__((ext_vector_type(16))) __bf16 bf16x16;
typedef __attribute__((ext_vector_type(8)))  __bf16 bf16x8;
typedef __attribute__((ext_vector_type(8)))  float  floatx8;

#define SHUF16(a, b) __builtin_shufflevector(a, b, 0, 1, 2, 3, 4, 5, 6, 7, 8, 9, 10, 11, 12, 13, 14, 15)

// =====================================================================
// Fused GEMM: C[M x 128] = act( A[M x K] @ W[128 x K]^T + bias )
// bf16 WMMA 16x16x32, fp32 accumulate. act: 0 = none, 1 = relu
// LDS tiles stored row-per-matrix-row so each lane's fragment is two
// contiguous 16B chunks -> ds_load_b128 instead of scalar u16 gathers.
// =====================================================================
__global__ __launch_bounds__(128)
void gemm_bias_act(const float* __restrict__ A, const float* __restrict__ W,
                   const float* __restrict__ bias, float* __restrict__ C,
                   int M, int K, int act) {
  __shared__ __attribute__((aligned(16))) __bf16 As[BM][BKP];  // [row][k]
  __shared__ __attribute__((aligned(16))) __bf16 Bs[BN][BKP];  // [n][k] = W[n][k0+k]

  const int tid  = threadIdx.x;
  const int wave = tid >> 5;
  const int lane = tid & 31;
  const int half = lane >> 4;     // 0: lanes 0-15, 1: lanes 16-31
  const int lm   = lane & 15;
  const int rowBase = blockIdx.x * BM;
  const int waveRow = rowBase + wave * 16;
  const int arow = wave * 16 + lm;

  floatx8 acc[8] = {};

  const int ksteps = (K + BK - 1) / BK;
  for (int ks = 0; ks < ksteps; ++ks) {
    const int k0 = ks * BK;
    // ---- stage A tile: 64 rows x 4 chunks of 8 (256 tasks, 2/thread) ----
    for (int t = tid; t < BM * (BK / 8); t += 128) {
      int r = t >> 2, c0 = (t & 3) << 3;
      int gr = rowBase + r;
      bf16x8 v;
#pragma unroll
      for (int j = 0; j < 8; ++j) {
        int gc = k0 + c0 + j;
        float f = (gr < M && gc < K) ? A[(size_t)gr * K + gc] : 0.0f;
        v[j] = (__bf16)f;
      }
      *(bf16x8*)&As[r][c0] = v;   // single ds_store_b128
    }
    // ---- stage B tile: 128 rows x 4 chunks of 8 (512 tasks, 4/thread) ----
    for (int t = tid; t < BN * (BK / 8); t += 128) {
      int n = t >> 2, c0 = (t & 3) << 3;
      bf16x8 v;
#pragma unroll
      for (int j = 0; j < 8; ++j) {
        int gc = k0 + c0 + j;
        float f = (gc < K) ? W[(size_t)n * K + gc] : 0.0f;
        v[j] = (__bf16)f;
      }
      *(bf16x8*)&Bs[n][c0] = v;   // single ds_store_b128
    }
    __syncthreads();

    // A fragment: elements 0-7 -> K = half*8..half*8+7 ; 8-15 -> K = 16+half*8..+7
    bf16x8 a0 = *(const bf16x8*)&As[arow][half << 3];
    bf16x8 a1 = *(const bf16x8*)&As[arow][16 + (half << 3)];
    bf16x16 afrag = SHUF16(a0, a1);

#pragma unroll
    for (int nt = 0; nt < 8; ++nt) {
      // B fragment: lane lm = col N, elements 0-7 -> K = half*8..half*8+7 ; 8-15 -> K = 16+half*8..+7
      const int brow = nt * 16 + lm;
      bf16x8 b0 = *(const bf16x8*)&Bs[brow][half << 3];
      bf16x8 b1 = *(const bf16x8*)&Bs[brow][16 + (half << 3)];
      bf16x16 bfrag = SHUF16(b0, b1);
      acc[nt] = __builtin_amdgcn_wmma_f32_16x16x32_bf16(
          false, afrag, false, bfrag, (short)0, acc[nt], false, false);
    }
    __syncthreads();
  }

  // D layout: VGPR r -> row r + 8*half, col = lm
#pragma unroll
  for (int nt = 0; nt < 8; ++nt) {
    int n  = nt * 16 + lm;
    float bv = bias ? bias[n] : 0.0f;
#pragma unroll
    for (int r = 0; r < 8; ++r) {
      int m = waveRow + r + (half << 3);
      if (m < M) {
        float v = acc[nt][r] + bv;
        if (act == 1) v = fmaxf(v, 0.0f);
        C[(size_t)m * BN + n] = v;
      }
    }
  }
}

// =====================================================================
// GCN propagation helpers
// =====================================================================
__global__ void fill1_f(float* p, int n) {
  int i = blockIdx.x * blockDim.x + threadIdx.x;
  if (i < n) p[i] = 1.0f;
}
__global__ void zero_f(float* p, size_t n) {
  size_t i = (size_t)blockIdx.x * blockDim.x + threadIdx.x;
  if (i < n) p[i] = 0.0f;
}
__global__ void deg_edges(const int* __restrict__ dst, float* __restrict__ deg, int E) {
  int i = blockIdx.x * blockDim.x + threadIdx.x;
  if (i < E) atomicAdd(&deg[dst[i]], 1.0f);
}
__global__ void rsqrt_f(float* p, int n) {
  int i = blockIdx.x * blockDim.x + threadIdx.x;
  if (i < n) p[i] = rsqrtf(p[i]);
}
// per edge: agg[dst] += h[src] * dinv[src]*dinv[dst]  (32 threads x float4 per edge)
__global__ void edge_scatter(const float* __restrict__ h, const int* __restrict__ src,
                             const int* __restrict__ dst, const float* __restrict__ dinv,
                             float* __restrict__ agg, int E) {
  int t = blockIdx.x * blockDim.x + threadIdx.x;
  int e = t >> 5, j = t & 31;
  if (e >= E) return;
  int s = src[e], d = dst[e];
  float c = dinv[s] * dinv[d];
  float4 v = ((const float4*)(h + (size_t)s * HD))[j];
  float* ap = agg + (size_t)d * HD + j * 4;
  atomicAdd(ap + 0, v.x * c);
  atomicAdd(ap + 1, v.y * c);
  atomicAdd(ap + 2, v.z * c);
  atomicAdd(ap + 3, v.w * c);
}
// out = act( agg + dense*dinv^2 + bias )
__global__ void gcn_finalize(const float* __restrict__ dense, const float* __restrict__ agg,
                             const float* __restrict__ dinv, const float* __restrict__ bias,
                             float* __restrict__ out, int N, int act) {
  size_t idx = (size_t)blockIdx.x * blockDim.x + threadIdx.x;
  if (idx >= (size_t)N * HD) return;
  int node = (int)(idx >> 7), f = (int)(idx & 127);
  float di = dinv[node];
  float v = agg[idx] + dense[idx] * di * di + bias[f];
  if (act == 1) v = fmaxf(v, 0.0f);
  out[idx] = v;
}

// =====================================================================
// Multi-head attention: Out[b,q,:] = Qin[b,q,:] + softmax(QK^T/sqrt(128)) V
// one block per (graph, head); K/V head slices staged in LDS
// qPerB == 0 -> Qin is a single [nq, HD] matrix broadcast over graphs
// =====================================================================
__global__ __launch_bounds__(256)
void attn_kernel(const float* __restrict__ Qin, const float* __restrict__ Kin,
                 const float* __restrict__ Vin, float* __restrict__ Out,
                 int nq, int nk, int qPerB) {
  __shared__ float Ksh[NPG * DH];
  __shared__ float Vsh[NPG * DH];
  __shared__ float Ash[256];
  __shared__ float red[256];
  __shared__ float part[16 * 16];

  const int b = blockIdx.x, hg = blockIdx.y, tid = threadIdx.x;
  const float scale = rsqrtf((float)HD);
  const float* Kb = Kin + (size_t)b * nk * HD;
  const float* Vb = Vin + (size_t)b * nk * HD;
  const float* Qb = Qin + (qPerB ? (size_t)b * nq * HD : 0);

  for (int i = tid; i < nk * DH; i += 256) {
    int kk = i >> 4, d = i & 15;
    Ksh[i] = Kb[(size_t)kk * HD + hg * DH + d];
    Vsh[i] = Vb[(size_t)kk * HD + hg * DH + d];
  }
  __syncthreads();

  for (int q = 0; q < nq; ++q) {
    const float* qv = Qb + (size_t)q * HD + hg * DH;
    float s = -1e30f;
    if (tid < nk) {
      float a = 0.0f;
#pragma unroll
      for (int d = 0; d < DH; ++d) a += qv[d] * Ksh[tid * DH + d];
      s = a * scale;
    }
    red[tid] = s;
    __syncthreads();
    for (int off = 128; off > 0; off >>= 1) {
      if (tid < off) red[tid] = fmaxf(red[tid], red[tid + off]);
      __syncthreads();
    }
    float mx = red[0];
    __syncthreads();
    float ex = (tid < nk) ? __expf(s - mx) : 0.0f;
    Ash[tid] = ex;
    red[tid] = ex;
    __syncthreads();
    for (int off = 128; off > 0; off >>= 1) {
      if (tid < off) red[tid] += red[tid + off];
      __syncthreads();
    }
    float inv = 1.0f / red[0];
    __syncthreads();

    int d = tid & 15, g = tid >> 4;
    float p = 0.0f;
    for (int kk = g; kk < nk; kk += 16) p += Ash[kk] * Vsh[kk * DH + d];
    part[g * 16 + d] = p;
    __syncthreads();
    if (tid < 16) {
      float o = 0.0f;
#pragma unroll
      for (int g2 = 0; g2 < 16; ++g2) o += part[g2 * 16 + tid];
      Out[((size_t)b * nq + q) * HD + hg * DH + tid] = qv[tid] + o * inv;
    }
    __syncthreads();
  }
}

// out = base + relu(t)
__global__ void add_relu(const float* __restrict__ base, const float* __restrict__ t,
                         float* __restrict__ out, size_t n) {
  size_t i = (size_t)blockIdx.x * blockDim.x + threadIdx.x;
  if (i < n) out[i] = base[i] + fmaxf(t[i], 0.0f);
}

// cat[b, 0:128] = a[b], cat[b, 128:256] = hv (broadcast)
__global__ void concat_herg(const float* __restrict__ a, const float* __restrict__ hv,
                            float* __restrict__ out, int B) {
  int i = blockIdx.x * blockDim.x + threadIdx.x;
  if (i < B * 256) {
    int b = i >> 8, f = i & 255;
    out[i] = (f < 128) ? a[b * 128 + f] : hv[f - 128];
  }
}

// d_out[g] = sigmoid( dot(x[g,:], w) + b )
__global__ void sig_head(const float* __restrict__ x, const float* __restrict__ w,
                         const float* __restrict__ bias, float* __restrict__ out, int B) {
  int g = blockIdx.x * blockDim.x + threadIdx.x;
  if (g < B) {
    float s = bias[0];
#pragma unroll 8
    for (int f = 0; f < HD; ++f) s += x[(size_t)g * HD + f] * w[f];
    out[g] = 1.0f / (1.0f + __expf(-s));
  }
}

// =====================================================================
// driver
// =====================================================================
extern "C" void kernel_launch(void* const* d_in, const int* in_sizes, int n_in,
                              void* d_out, int out_size, void* d_ws, size_t ws_size,
                              hipStream_t stream) {
  (void)in_sizes; (void)n_in; (void)out_size; (void)ws_size;
  const float* herg_em   = (const float*)d_in[0];
  const float* x         = (const float*)d_in[1];
  // d_in[2] edge_attr, d_in[3] mol_rep: unused by reference
  const float* convW[4]  = {(const float*)d_in[4], (const float*)d_in[6],
                            (const float*)d_in[8], (const float*)d_in[10]};
  const float* convB[4]  = {(const float*)d_in[5], (const float*)d_in[7],
                            (const float*)d_in[9], (const float*)d_in[11]};
  const float* lin1_W = (const float*)d_in[12]; const float* lin1_b = (const float*)d_in[13];
  const float* lin2_W = (const float*)d_in[14]; const float* lin2_b = (const float*)d_in[15];
  const float* S1     = (const float*)d_in[16];
  const float* p1_fq_W = (const float*)d_in[17]; const float* p1_fq_b = (const float*)d_in[18];
  const float* p1_k_W  = (const float*)d_in[19]; const float* p1_k_b  = (const float*)d_in[20];
  const float* p1_v_W  = (const float*)d_in[21]; const float* p1_v_b  = (const float*)d_in[22];
  const float* p1_fo_W = (const float*)d_in[23]; const float* p1_fo_b = (const float*)d_in[24];
  const float* sb_fq_W = (const float*)d_in[25]; const float* sb_fq_b = (const float*)d_in[26];
  const float* sb_k_W  = (const float*)d_in[27]; const float* sb_k_b  = (const float*)d_in[28];
  const float* sb_v_W  = (const float*)d_in[29]; const float* sb_v_b  = (const float*)d_in[30];
  const float* sb_fo_W = (const float*)d_in[31]; const float* sb_fo_b = (const float*)d_in[32];
  const float* p2_fq_W = (const float*)d_in[33]; const float* p2_fq_b = (const float*)d_in[34];
  const float* p2_k_W  = (const float*)d_in[35]; const float* p2_k_b  = (const float*)d_in[36];
  const float* p2_v_W  = (const float*)d_in[37]; const float* p2_v_b  = (const float*)d_in[38];
  const float* p2_fo_W = (const float*)d_in[39]; const float* p2_fo_b = (const float*)d_in[40];
  const float* S2      = (const float*)d_in[41];
  const float* hlin_W  = (const float*)d_in[42]; const float* hlin_b = (const float*)d_in[43];
  const float* hout_W  = (const float*)d_in[44]; const float* hout_b = (const float*)d_in[45];
  const float* fc0_W   = (const float*)d_in[46]; const float* fc0_b  = (const float*)d_in[47];
  const float* sig_W   = (const float*)d_in[48]; const float* sig_b  = (const float*)d_in[49];
  const int*   eidx    = (const int*)d_in[50];
  const int*   src = eidx;
  const int*   dst = eidx + N_EDGES;
  float* out = (float*)d_out;

  // workspace layout
  float* ws = (float*)d_ws;
  const size_t NH = (size_t)N_NODES * HD;       // 12.8M floats
  float* P0 = ws;
  float* P1 = ws + NH;
  float* P2 = ws + 2 * NH;
  float* P3 = ws + 3 * NH;
  float* dinv = ws + 4 * NH;                    // N floats
  float* Qb1  = dinv + N_NODES;                 // 75*128
  float* Qb2  = Qb1 + KSEEDS * HD;              // 128
  float* hv   = Qb2 + HD;                       // 128
  const size_t XS = (size_t)N_GRAPHS * KSEEDS * HD;  // 4.8M floats

  auto gemm = [&](const float* A, const float* W, const float* b, float* C,
                  int M, int K, int act) {
    gemm_bias_act<<<dim3((M + BM - 1) / BM), dim3(128), 0, stream>>>(A, W, b, C, M, K, act);
  };
  auto propagate = [&](const float* dense, float* agg) {
    zero_f<<<(unsigned)((NH + 255) / 256), 256, 0, stream>>>(agg, NH);
    edge_scatter<<<(N_EDGES * 32 + 255) / 256, 256, 0, stream>>>(dense, src, dst, dinv, agg, N_EDGES);
  };
  const int NB_NH = (int)((NH + 255) / 256);

  // ---- degree: deg = 1 + in-degree; dinv = deg^-1/2 ----
  fill1_f<<<(N_NODES + 255) / 256, 256, 0, stream>>>(dinv, N_NODES);
  deg_edges<<<(N_EDGES + 255) / 256, 256, 0, stream>>>(dst, dinv, N_EDGES);
  rsqrt_f<<<(N_NODES + 255) / 256, 256, 0, stream>>>(dinv, N_NODES);

  // ---- 4x (GCNConv + ReLU) ----
  gemm(x, convW[0], nullptr, P1, N_NODES, 37, 0);
  propagate(P1, P2);
  gcn_finalize<<<NB_NH, 256, 0, stream>>>(P1, P2, dinv, convB[0], P0, N_NODES, 1);
  for (int l = 1; l < 4; ++l) {
    gemm(P0, convW[l], nullptr, P1, N_NODES, HD, 0);
    propagate(P1, P2);
    gcn_finalize<<<NB_NH, 256, 0, stream>>>(P1, P2, dinv, convB[l], P0, N_NODES, 1);
  }

  // ---- hx = h @ lin1 + b -> P3 ----
  gemm(P0, lin1_W, lin1_b, P3, N_NODES, HD, 0);

  // ---- GMPool_G: K = gcn(hx; p1_k) -> P0, V = gcn(hx; p1_v) -> P1 ----
  gemm(P3, p1_k_W, nullptr, P1, N_NODES, HD, 0);
  propagate(P1, P2);
  gcn_finalize<<<NB_NH, 256, 0, stream>>>(P1, P2, dinv, p1_k_b, P0, N_NODES, 0);
  gemm(P3, p1_v_W, nullptr, P1, N_NODES, HD, 0);
  propagate(P1, P2);
  gcn_finalize<<<NB_NH, 256, 0, stream>>>(P1, P2, dinv, p1_v_b, P1, N_NODES, 0);

  // Q = S1 @ fq^T + b (shared across graphs)
  gemm(S1, p1_fq_W, p1_fq_b, Qb1, KSEEDS, HD, 0);
  attn_kernel<<<dim3(N_GRAPHS, NHEADS), 256, 0, stream>>>(Qb1, P0, P1, P2, KSEEDS, NPG, 0);
  gemm(P2, p1_fo_W, p1_fo_b, P3, N_GRAPHS * KSEEDS, HD, 0);
  add_relu<<<(unsigned)((XS + 255) / 256), 256, 0, stream>>>(P2, P3, P2, XS);  // xb -> P2

  // ---- SAB ----
  gemm(P2, sb_fq_W, sb_fq_b, P0, N_GRAPHS * KSEEDS, HD, 0);       // Q -> P0[0:XS)
  gemm(P2, sb_k_W, sb_k_b, P1, N_GRAPHS * KSEEDS, HD, 0);         // K -> P1
  gemm(P2, sb_v_W, sb_v_b, P3, N_GRAPHS * KSEEDS, HD, 0);         // V -> P3
  attn_kernel<<<dim3(N_GRAPHS, NHEADS), 256, 0, stream>>>(P0, P1, P3, P0 + XS, KSEEDS, KSEEDS, 1);
  gemm(P0 + XS, sb_fo_W, sb_fo_b, P3, N_GRAPHS * KSEEDS, HD, 0);
  add_relu<<<(unsigned)((XS + 255) / 256), 256, 0, stream>>>(P0 + XS, P3, P2, XS);  // xb -> P2

  // ---- GMPool_I ----
  gemm(S2, p2_fq_W, p2_fq_b, Qb2, 1, HD, 0);
  gemm(P2, p2_k_W, p2_k_b, P0, N_GRAPHS * KSEEDS, HD, 0);
  gemm(P2, p2_v_W, p2_v_b, P1, N_GRAPHS * KSEEDS, HD, 0);
  attn_kernel<<<dim3(N_GRAPHS, NHEADS), 256, 0, stream>>>(Qb2, P0, P1, P3, 1, KSEEDS, 0);
  gemm(P3, p2_fo_W, p2_fo_b, P0, N_GRAPHS, HD, 0);
  add_relu<<<(N_GRAPHS * HD + 255) / 256, 256, 0, stream>>>(P3, P0, P1, (size_t)N_GRAPHS * HD);

  // ---- out = xb2 @ lin2 + b -> P2 [500,128] ----
  gemm(P1, lin2_W, lin2_b, P2, N_GRAPHS, HD, 0);

  // ---- hERG branch + head ----
  gemm(herg_em, hlin_W, hlin_b, hv, 1, 1280, 1);                  // relu([1,1280]@[1280,128])
  concat_herg<<<(N_GRAPHS * 256 + 255) / 256, 256, 0, stream>>>(P2, hv, P0, N_GRAPHS);
  gemm(P0, hout_W, hout_b, P1, N_GRAPHS, 256, 1);                 // relu
  gemm(P1, fc0_W, fc0_b, P2, N_GRAPHS, HD, 1);                    // relu
  sig_head<<<(N_GRAPHS + 255) / 256, 256, 0, stream>>>(P2, sig_W, sig_b, out, N_GRAPHS);
}
